// GatedGCN_LSPELayer_22789096473264
// MI455X (gfx1250) — matrix operand order repaired and checked by reading, not compiled, not run on hardware
//
#include <hip/hip_runtime.h>

// GatedGCN-LSPE layer for MI455X (gfx1250, wave32, WMMA bf16 16x16x32).
// D = 128 fixed. Multi-pass because BatchNorm needs global batch statistics.

typedef __attribute__((ext_vector_type(16))) __bf16 v16bf;
typedef __attribute__((ext_vector_type(8)))  float  v8f;

#define DD 128
#define WMMA_BF16(A, B, C) \
  __builtin_amdgcn_wmma_f32_16x16x32_bf16(false, (A), false, (B), (short)0, (C), false, false)

// ---- Fragment loaders (ISA 7.12.2 layouts, wave32) ----
// A (16x32 bf16): lane = n + 16*half, row M = n.
//   half 0: K = {0..7} in elems 0..7, {16..23} in elems 8..15
//   half 1: K = {8..15} and {24..31}
__device__ __forceinline__ v16bf a_frag(const float* __restrict__ row, int k0, int half) {
  v16bf a;
  const int c0 = k0 + half * 8;
  const int c1 = k0 + 16 + half * 8;
#pragma unroll
  for (int i = 0; i < 8; ++i) {
    a[i]     = (__bf16)row[c0 + i];
    a[8 + i] = (__bf16)row[c1 + i];
  }
  return a;
}

__device__ __forceinline__ v16bf a_frag3(const float* __restrict__ r0, const float* __restrict__ r1,
                                         const float* __restrict__ r2, int k0, int half) {
  v16bf a;
  const int c0 = k0 + half * 8;
  const int c1 = k0 + 16 + half * 8;
#pragma unroll
  for (int i = 0; i < 8; ++i) {
    a[i]     = (__bf16)(r0[c0 + i] + r1[c0 + i] + r2[c0 + i]);
    a[8 + i] = (__bf16)(r0[c1 + i] + r1[c1 + i] + r2[c1 + i]);
  }
  return a;
}

// B (32x16 bf16): lane = n + 16*half holds column N = n, K = k0 + half*8 + {0..7} and k0 + 16 + half*8 + {0..7}.
// Weights are pre-transposed to [N][K] bf16 so this is two contiguous 16B reads.
__device__ __forceinline__ v16bf b_frag(const __bf16* __restrict__ wt, int ncol, int k0, int half) {
  const __bf16* cp = wt + (size_t)ncol * DD + k0 + half * 8;
  v16bf b;
#pragma unroll
  for (int i = 0; i < 8; ++i) { b[i] = cp[i]; b[8 + i] = cp[16 + i]; }
  return b;
}

// ---- Kernel 1: transpose+convert 7 DxD weight blocks fp32[K][N] -> bf16[N][K] ----
// blocks: 0=W_lin 1=W_hps[:D] 2=W_hps[D:] 3=W_hpr[:D] 4=W_hpr[D:] 5=W_p1 6=W_p2
__global__ __launch_bounds__(128) void prep_weights(
    const float* __restrict__ Wlin, const float* __restrict__ Whps,
    const float* __restrict__ Whpr, const float* __restrict__ Wp1,
    const float* __restrict__ Wp2, __bf16* __restrict__ out) {
  const int ncol = blockIdx.x;   // 0..127 (N)
  const int matid = blockIdx.y;  // 0..6
  const int k = threadIdx.x;     // 0..127 (K)
  const float* src;
  int row = k;
  switch (matid) {
    case 0: src = Wlin; break;
    case 1: src = Whps; break;
    case 2: src = Whps; row = 128 + k; break;
    case 3: src = Whpr; break;
    case 4: src = Whpr; row = 128 + k; break;
    case 5: src = Wp1; break;
    default: src = Wp2; break;
  }
  out[(size_t)matid * 16384 + (size_t)ncol * DD + k] = (__bf16)src[(size_t)row * DD + ncol];
}

// ---- Kernel 2: per-edge GEMMs + eta + atomic segment sums ----
// One wave = one 16-edge tile. 4 waves / 128-thread block.
__global__ __launch_bounds__(128) void pass_a(
    const float* __restrict__ h, const float* __restrict__ e, const float* __restrict__ p,
    const int* __restrict__ send, const int* __restrict__ rec,
    const float* __restrict__ b_lin, const float* __restrict__ b_hps,
    const float* __restrict__ b_hpr, const float* __restrict__ b_p2,
    const __bf16* __restrict__ wts,
    float* __restrict__ eta_ws, float* __restrict__ hps_ws,
    float* __restrict__ aggh, float* __restrict__ aggp, int E) {
  const int wave = threadIdx.x >> 5;
  const int lane = threadIdx.x & 31;
  const int tile = blockIdx.x * 4 + wave;
  if (tile * 16 >= E) return;  // wave-uniform
  const int ebase = tile * 16;
  const int n = lane & 15;
  const int half = lane >> 4;

  int my_edge = ebase + n;
  if (my_edge >= E) my_edge = E - 1;  // clamp loads; stores guarded below
  const int s_a = send[my_edge];
  const int r_a = rec[my_edge];
  const float* hs_row = h + (size_t)s_a * DD;
  const float* hr_row = h + (size_t)r_a * DD;
  const float* ps_row = p + (size_t)s_a * DD;
  const float* pr_row = p + (size_t)r_a * DD;
  const float* e_row  = e + (size_t)my_edge * DD;

  const __bf16* w_lin  = wts;
  const __bf16* w_hps0 = wts + 1 * 16384;
  const __bf16* w_hps1 = wts + 2 * 16384;
  const __bf16* w_hpr0 = wts + 3 * 16384;
  const __bf16* w_hpr1 = wts + 4 * 16384;
  const __bf16* w_p2   = wts + 6 * 16384;

  // C/D layout rows for this lane: M = v + 8*half
  int erow[8], r_c[8];
#pragma unroll
  for (int v = 0; v < 8; ++v) {
    const int rr = ebase + v + 8 * half;
    erow[v] = (rr < E) ? rr : -1;
    r_c[v] = rec[(rr < E) ? rr : 0];
  }

  // ================= eta = sigmoid((h_s+h_r+e) @ W_lin + 3*b_lin) =================
  v8f etan[8];
#pragma unroll
  for (int nt = 0; nt < 8; ++nt) {
    const float b3 = 3.0f * b_lin[nt * 16 + n];
    v8f c;
#pragma unroll
    for (int v = 0; v < 8; ++v) c[v] = b3;
    etan[nt] = c;
  }
#pragma unroll
  for (int k0 = 0; k0 < DD; k0 += 32) {
    const v16bf a = a_frag3(hs_row, hr_row, e_row, k0, half);
#pragma unroll
    for (int nt = 0; nt < 8; ++nt) {
      const v16bf b = b_frag(w_lin, nt * 16 + n, k0, half);
      etan[nt] = WMMA_BF16(a, b, etan[nt]);
    }
  }
  float rs[8];
#pragma unroll
  for (int v = 0; v < 8; ++v) rs[v] = 0.0f;
#pragma unroll
  for (int nt = 0; nt < 8; ++nt) {
#pragma unroll
    for (int v = 0; v < 8; ++v) {
      const float ev = 1.0f / (1.0f + expf(-etan[nt][v]));
      etan[nt][v] = ev;
      rs[v] += ev;
      if (erow[v] >= 0) eta_ws[(size_t)erow[v] * DD + nt * 16 + n] = ev;
    }
  }
  // row-sum over N: reduce across the 16 lanes of each half (masks 1,2,4,8 stay in-half)
#pragma unroll
  for (int v = 0; v < 8; ++v) {
    float s = rs[v];
    s += __shfl_xor(s, 1, 32);
    s += __shfl_xor(s, 2, 32);
    s += __shfl_xor(s, 4, 32);
    s += __shfl_xor(s, 8, 32);
    const float inv = 1.0f / s;
#pragma unroll
    for (int nt = 0; nt < 8; ++nt) etan[nt][v] *= inv;  // etan := eta_new
  }

  // ================= hp_send = [h_s|p_s] @ W_hps + b_hps  (store) =================
  v8f acc[8];
#pragma unroll
  for (int nt = 0; nt < 8; ++nt) {
    const float bb = b_hps[nt * 16 + n];
    v8f c;
#pragma unroll
    for (int v = 0; v < 8; ++v) c[v] = bb;
    acc[nt] = c;
  }
#pragma unroll
  for (int k0 = 0; k0 < DD; k0 += 32) {
    const v16bf a = a_frag(hs_row, k0, half);
#pragma unroll
    for (int nt = 0; nt < 8; ++nt)
      acc[nt] = WMMA_BF16(a, b_frag(w_hps0, nt * 16 + n, k0, half), acc[nt]);
  }
#pragma unroll
  for (int k0 = 0; k0 < DD; k0 += 32) {
    const v16bf a = a_frag(ps_row, k0, half);
#pragma unroll
    for (int nt = 0; nt < 8; ++nt)
      acc[nt] = WMMA_BF16(a, b_frag(w_hps1, nt * 16 + n, k0, half), acc[nt]);
  }
#pragma unroll
  for (int nt = 0; nt < 8; ++nt)
#pragma unroll
    for (int v = 0; v < 8; ++v)
      if (erow[v] >= 0) hps_ws[(size_t)erow[v] * DD + nt * 16 + n] = acc[nt][v];

  // ====== hp_rec = [h_r|p_r] @ W_hpr + b_hpr ; agg_h[rec] += hp_rec * eta_new ======
#pragma unroll
  for (int nt = 0; nt < 8; ++nt) {
    const float bb = b_hpr[nt * 16 + n];
    v8f c;
#pragma unroll
    for (int v = 0; v < 8; ++v) c[v] = bb;
    acc[nt] = c;
  }
#pragma unroll
  for (int k0 = 0; k0 < DD; k0 += 32) {
    const v16bf a = a_frag(hr_row, k0, half);
#pragma unroll
    for (int nt = 0; nt < 8; ++nt)
      acc[nt] = WMMA_BF16(a, b_frag(w_hpr0, nt * 16 + n, k0, half), acc[nt]);
  }
#pragma unroll
  for (int k0 = 0; k0 < DD; k0 += 32) {
    const v16bf a = a_frag(pr_row, k0, half);
#pragma unroll
    for (int nt = 0; nt < 8; ++nt)
      acc[nt] = WMMA_BF16(a, b_frag(w_hpr1, nt * 16 + n, k0, half), acc[nt]);
  }
#pragma unroll
  for (int nt = 0; nt < 8; ++nt)
#pragma unroll
    for (int v = 0; v < 8; ++v)
      if (erow[v] >= 0)
        atomicAdd(&aggh[(size_t)r_c[v] * DD + nt * 16 + n], acc[nt][v] * etan[nt][v]);

  // ====== linear_p = p_r @ W_p2 + b_p2 ; agg_p[rec] += linear_p * eta_new ======
#pragma unroll
  for (int nt = 0; nt < 8; ++nt) {
    const float bb = b_p2[nt * 16 + n];
    v8f c;
#pragma unroll
    for (int v = 0; v < 8; ++v) c[v] = bb;
    acc[nt] = c;
  }
#pragma unroll
  for (int k0 = 0; k0 < DD; k0 += 32) {
    const v16bf a = a_frag(pr_row, k0, half);
#pragma unroll
    for (int nt = 0; nt < 8; ++nt)
      acc[nt] = WMMA_BF16(a, b_frag(w_p2, nt * 16 + n, k0, half), acc[nt]);
  }
#pragma unroll
  for (int nt = 0; nt < 8; ++nt)
#pragma unroll
    for (int v = 0; v < 8; ++v)
      if (erow[v] >= 0)
        atomicAdd(&aggp[(size_t)r_c[v] * DD + nt * 16 + n], acc[nt][v] * etan[nt][v]);
}

// ---- Kernel 3: x = hp_send + agg_h (in place) + per-column sum/sumsq stats ----
__global__ __launch_bounds__(128) void pass_b(
    float* __restrict__ x, const float* __restrict__ aggh,
    const float* __restrict__ eta, float* __restrict__ stats, int E) {
  const int c = threadIdx.x;  // column 0..127, coalesced across threads
  const int r0 = blockIdx.x * 1024;
  int r1 = r0 + 1024;
  if (r1 > E) r1 = E;
  float sx = 0.f, sxx = 0.f, se = 0.f, see = 0.f;
  for (int r = r0; r < r1; ++r) {
    const size_t idx = (size_t)r * DD + c;
    const float xv = x[idx] + aggh[idx];
    x[idx] = xv;
    sx += xv;
    sxx += xv * xv;
    const float ev = eta[idx];
    se += ev;
    see += ev * ev;
  }
  atomicAdd(&stats[c], sx);
  atomicAdd(&stats[128 + c], sxx);
  atomicAdd(&stats[256 + c], se);
  atomicAdd(&stats[384 + c], see);
}

// ---- Kernel 4: sums -> mean / inv_std (in place) ----
__global__ __launch_bounds__(128) void finalize_stats(float* __restrict__ stats, int E) {
  const int c = threadIdx.x;
  const float invE = 1.0f / (float)E;
  const float mx = stats[c] * invE;
  const float vx = stats[128 + c] * invE - mx * mx;
  const float me = stats[256 + c] * invE;
  const float ve = stats[384 + c] * invE - me * me;
  stats[c] = mx;
  stats[128 + c] = rsqrtf(vx + 1e-5f);
  stats[256 + c] = me;
  stats[384 + c] = rsqrtf(ve + 1e-5f);
}

// ---- Kernel 5: p_s @ W_p1 (WMMA), BN+relu residuals, scatter outputs ----
__global__ __launch_bounds__(128) void pass_c(
    const float* __restrict__ h, const float* __restrict__ e, const float* __restrict__ p,
    const int* __restrict__ send,
    const float* __restrict__ b_p1, const float* __restrict__ gamma, const float* __restrict__ beta,
    const __bf16* __restrict__ wts, const float* __restrict__ x_ws,
    const float* __restrict__ eta_ws, const float* __restrict__ aggp,
    const float* __restrict__ stats,
    float* __restrict__ out_h, float* __restrict__ out_e, float* __restrict__ out_p, int E) {
  const int wave = threadIdx.x >> 5;
  const int lane = threadIdx.x & 31;
  const int tile = blockIdx.x * 4 + wave;
  if (tile * 16 >= E) return;
  const int ebase = tile * 16;
  const int n = lane & 15;
  const int half = lane >> 4;

  int my_edge = ebase + n;
  if (my_edge >= E) my_edge = E - 1;
  const int s_a = send[my_edge];
  const float* ps_row = p + (size_t)s_a * DD;
  const __bf16* w_p1 = wts + 5 * 16384;

  int erow[8], s_c[8];
#pragma unroll
  for (int v = 0; v < 8; ++v) {
    const int rr = ebase + v + 8 * half;
    erow[v] = (rr < E) ? rr : -1;
    s_c[v] = send[(rr < E) ? rr : 0];
  }

  // acc = p_s @ W_p1 + b_p1
  v8f acc[8];
#pragma unroll
  for (int nt = 0; nt < 8; ++nt) {
    const float bb = b_p1[nt * 16 + n];
    v8f c;
#pragma unroll
    for (int v = 0; v < 8; ++v) c[v] = bb;
    acc[nt] = c;
  }
#pragma unroll
  for (int k0 = 0; k0 < DD; k0 += 32) {
    const v16bf a = a_frag(ps_row, k0, half);
#pragma unroll
    for (int nt = 0; nt < 8; ++nt)
      acc[nt] = WMMA_BF16(a, b_frag(w_p1, nt * 16 + n, k0, half), acc[nt]);
  }

#pragma unroll
  for (int nt = 0; nt < 8; ++nt) {
    const int col = nt * 16 + n;
    const float gm = gamma[col], bt = beta[col];
    const float mx = stats[col], ix = stats[128 + col];
    const float me = stats[256 + col], ie = stats[384 + col];
#pragma unroll
    for (int v = 0; v < 8; ++v) {
      if (erow[v] < 0) continue;
      const size_t idx = (size_t)erow[v] * DD + col;
      const size_t sidx = (size_t)s_c[v] * DD + col;
      // h_out[send] = h_s + relu(bn(hp_send + agg_h))
      float bnx = gm * (x_ws[idx] - mx) * ix + bt;
      bnx = bnx > 0.f ? bnx : 0.f;
      out_h[sidx] = h[sidx] + bnx;
      // e_out = e + relu(bn(eta))
      float bne = gm * (eta_ws[idx] - me) * ie + bt;
      bne = bne > 0.f ? bne : 0.f;
      out_e[idx] = e[idx] + bne;
      // p_out[send] = p_s + tanh(p_s @ W_p1 + b_p1 + agg_p)
      const float pv = acc[nt][v] + aggp[idx];
      out_p[sidx] = p[sidx] + tanhf(pv);
    }
  }
}

extern "C" void kernel_launch(void* const* d_in, const int* in_sizes, int n_in,
                              void* d_out, int out_size, void* d_ws, size_t ws_size,
                              hipStream_t stream) {
  (void)n_in; (void)out_size; (void)ws_size;
  const float* h     = (const float*)d_in[0];
  const float* e     = (const float*)d_in[1];
  const float* p     = (const float*)d_in[2];
  const int*   ei    = (const int*)d_in[3];
  const float* W_lin = (const float*)d_in[4];
  const float* b_lin = (const float*)d_in[5];
  const float* W_hps = (const float*)d_in[6];
  const float* b_hps = (const float*)d_in[7];
  const float* W_hpr = (const float*)d_in[8];
  const float* b_hpr = (const float*)d_in[9];
  const float* W_p1  = (const float*)d_in[10];
  const float* b_p1  = (const float*)d_in[11];
  const float* W_p2  = (const float*)d_in[12];
  const float* b_p2  = (const float*)d_in[13];
  const float* gamma = (const float*)d_in[14];
  const float* beta  = (const float*)d_in[15];

  const int N = in_sizes[0] / DD;
  const int E = in_sizes[1] / DD;
  const int* send = ei;
  const int* rec  = ei + E;

  // Workspace layout (floats): eta | hp_send->x | agg_h | agg_p | stats(512) | bf16 weights
  float* ws = (float*)d_ws;
  const size_t mat = (size_t)E * DD;
  float* eta_ws = ws;
  float* x_ws   = ws + mat;
  float* aggh   = ws + 2 * mat;
  float* aggp   = ws + 3 * mat;
  float* stats  = ws + 4 * mat;
  __bf16* wts   = (__bf16*)(stats + 512);

  float* out_h = (float*)d_out;
  float* out_e = out_h + (size_t)N * DD;
  float* out_p = out_e + mat;

  // 1) weights -> bf16 [N][K]
  prep_weights<<<dim3(128, 7), 128, 0, stream>>>(W_lin, W_hps, W_hpr, W_p1, W_p2, wts);
  // 2) zero segment-sum buffers + stats
  hipMemsetAsync(aggh, 0, 2 * mat * sizeof(float), stream);  // aggh + aggp contiguous
  hipMemsetAsync(stats, 0, 512 * sizeof(float), stream);
  // 3) edge GEMMs + eta + atomic segment sums
  const int tiles = (E + 15) / 16;
  const int blocksA = (tiles + 3) / 4;
  pass_a<<<blocksA, 128, 0, stream>>>(h, e, p, send, rec, b_lin, b_hps, b_hpr, b_p2, wts,
                                      eta_ws, x_ws, aggh, aggp, E);
  // 4) x = hp_send + agg_h ; BN statistics
  const int blocksB = (E + 1023) / 1024;
  pass_b<<<blocksB, 128, 0, stream>>>(x_ws, aggh, eta_ws, stats, E);
  finalize_stats<<<1, 128, 0, stream>>>(stats, E);
  // 5) base copies for scatter outputs (capture-safe d2d async copies)
  hipMemcpyAsync(out_h, h, (size_t)N * DD * sizeof(float), hipMemcpyDeviceToDevice, stream);
  hipMemcpyAsync(out_p, p, (size_t)N * DD * sizeof(float), hipMemcpyDeviceToDevice, stream);
  // 6) final residuals + scatters
  pass_c<<<blocksA, 128, 0, stream>>>(h, e, p, send, b_p1, gamma, beta, wts, x_ws, eta_ws,
                                      aggp, stats, out_h, out_e, out_p, E);
}
